// TinySelectiveSSMBlock_89945205113386
// MI455X (gfx1250) — compile-verified, matrix operand order checked by reading, not run on hardware
//
#include <hip/hip_runtime.h>
#include <math.h>

// ---------- CDNA5 WMMA types ----------
typedef __attribute__((ext_vector_type(16))) __bf16 bf16x16;
typedef __attribute__((ext_vector_type(8)))  float  v8f;
typedef int v4i_gcc __attribute__((vector_size(16)));   // matches builtin param

#define Nb 32
#define Tt 512
#define Cc 768
#define Hh 1536
#define Mm (Nb * Tt)   // 16384 rows

// ---------- optional gfx1250 async global->LDS path ----------
#if defined(__has_builtin)
#if __has_builtin(__builtin_amdgcn_global_load_async_to_lds_b128) && \
    __has_builtin(__builtin_amdgcn_s_wait_asynccnt)
#define USE_ASYNC 1
#endif
#endif
#ifndef USE_ASYNC
#define USE_ASYNC 0
#endif

// fp32 -> bf16 round-to-nearest-even
static __device__ inline unsigned short f2bf(float x) {
    unsigned int u = __float_as_uint(x);
    unsigned int r = u + 0x7FFFu + ((u >> 16) & 1u);
    return (unsigned short)(r >> 16);
}

// copy 16 bytes global -> LDS (async DMA when available)
static __device__ inline void cp16(const unsigned short* g, unsigned short* l) {
#if USE_ASYNC
    __builtin_amdgcn_global_load_async_to_lds_b128(
        (__attribute__((address_space(1))) v4i_gcc*)(g),
        (__attribute__((address_space(3))) v4i_gcc*)(l), 0, 0);
#else
    *reinterpret_cast<uint4*>(l) = *reinterpret_cast<const uint4*>(g);
#endif
}
static __device__ inline void cp_wait_all() {
#if USE_ASYNC
    __builtin_amdgcn_s_wait_asynccnt(0);
#endif
}

// Load one 16x32 bf16 fragment from LDS (row-major, ldt elements per row).
// 16-bit A 16x32 layout: lane<16 -> K {0..7,16..23}, lane>=16 -> K {8..15,24..31}
static __device__ inline bf16x16 ld_frag(const unsigned short* smem, int ldt) {
    int lane = threadIdx.x & 31;
    int row  = lane & 15;
    int kh   = (lane >> 4) << 3;
    const unsigned short* p = smem + row * ldt + kh;
    union { bf16x16 v; uint4 q[2]; } f;
    f.q[0] = *reinterpret_cast<const uint4*>(p);
    f.q[1] = *reinterpret_cast<const uint4*>(p + 16);
    return f.v;
}

enum { EPI_OUT = 0, EPI_XDUAL = 1, EPI_GELU = 2, EPI_RESID_DUAL = 3, EPI_GATE = 4 };

// C = A(MxK, bf16) * B(NoutxK, bf16)^T, f32 accumulate, fused epilogue.
// Block tile 128x256x32, 256 threads = 8 waves (2x4), wave tile 64x64
// (16 WMMAs / 16 ds_load_b128 per wave per K step), LDS double-buffered
// with async global->LDS staging overlapping the WMMA stream.
template <int EPI>
__global__ __launch_bounds__(256) void gemm_bf16_wmma(
    const unsigned short* __restrict__ A, const unsigned short* __restrict__ B,
    float* __restrict__ out, unsigned short* __restrict__ outbf,
    float* __restrict__ out2, const float* __restrict__ bias,
    const float* __restrict__ resid, const float* __restrict__ sup,
    const float* __restrict__ wsg, const float* __restrict__ bsg,
    const float* __restrict__ xtbuf, int K, int Nout) {
    constexpr int BM = 128, BN = 256, BK = 32, LDT = BK + 8;  // 80B rows, 16B aligned
    __shared__ unsigned short sA[2][BM * LDT];  // 2 x 10 KiB
    __shared__ unsigned short sB[2][BN * LDT];  // 2 x 20 KiB

    const int tid  = threadIdx.x;
    const int m0   = blockIdx.x * BM;
    const int n0   = blockIdx.y * BN;
    const int wave = tid >> 5;
    const int wm   = wave >> 2;   // 0..1
    const int wn   = wave & 3;    // 0..3
    const int lane = tid & 31;

    v8f acc[4][4] = {};

    auto stage = [&](int k0, int pb) {
#pragma unroll
        for (int j = 0; j < 2; ++j) {                 // A: 512 x 16B chunks
            int c = tid + j * 256;
            int r = c >> 2, s = c & 3;
            cp16(A + (size_t)(m0 + r) * K + k0 + s * 8, &sA[pb][r * LDT + s * 8]);
        }
#pragma unroll
        for (int j = 0; j < 4; ++j) {                 // B: 1024 x 16B chunks
            int c = tid + j * 256;
            int r = c >> 2, s = c & 3;
            cp16(B + (size_t)(n0 + r) * K + k0 + s * 8, &sB[pb][r * LDT + s * 8]);
        }
    };

    stage(0, 0);
    cp_wait_all();
    __syncthreads();

    int p = 0;
    for (int k0 = 0; k0 < K; k0 += BK) {
        if (k0 + BK < K) stage(k0 + BK, p ^ 1);   // DMA next tile during compute
        bf16x16 bfr[4];
#pragma unroll
        for (int ni = 0; ni < 4; ++ni)
            bfr[ni] = ld_frag(&sB[p][(wn * 64 + ni * 16) * LDT], LDT);
#pragma unroll
        for (int mi = 0; mi < 4; ++mi) {
            bf16x16 afr = ld_frag(&sA[p][(wm * 64 + mi * 16) * LDT], LDT);
#pragma unroll
            for (int ni = 0; ni < 4; ++ni)
                acc[mi][ni] = __builtin_amdgcn_wmma_f32_16x16x32_bf16(
                    false, afr, false, bfr[ni], (short)0, acc[mi][ni], false, false);
        }
        cp_wait_all();
        __syncthreads();
        p ^= 1;
    }

    // epilogue (C/D: lane<16 -> M=r, lane>=16 -> M=r+8; N = lane&15)
    const int ln  = lane & 15;
    const int mhi = (lane >> 4) * 8;
#pragma unroll
    for (int mi = 0; mi < 4; ++mi) {
#pragma unroll
        for (int ni = 0; ni < 4; ++ni) {
            int tm = m0 + wm * 64 + mi * 16;
            int tn = n0 + wn * 64 + ni * 16 + ln;
#pragma unroll
            for (int r = 0; r < 8; ++r) {
                int m = tm + r + mhi;
                float v = acc[mi][ni][r];
                size_t o = (size_t)m * Nout + tn;
                if constexpr (EPI == EPI_OUT) {
                    out[o] = v;
                } else if constexpr (EPI == EPI_XDUAL) {
                    out[o] = v;                 // f32 for residual
                    outbf[o] = f2bf(v);         // bf16 for next GEMM's A
                } else if constexpr (EPI == EPI_GELU) {
                    v += bias[tn];
                    float g = 0.5f * v * (1.0f + erff(v * 0.70710678118654752f));
                    outbf[o] = f2bf(g);
                } else if constexpr (EPI == EPI_RESID_DUAL) {
                    v += bias[tn] + resid[o];
                    out[o] = v;
                    outbf[o] = f2bf(v);
                } else {  // EPI_GATE
                    float g1 = 1.0f / (1.0f + __expf(-(v + bias[tn])));
                    float g2 = 1.0f / (1.0f + __expf(-(sup[m] * wsg[tn] + bsg[tn])));
                    float g  = g1 * g2;
                    out[o]  = 1.0f - g;          // a_t (f32 for scan accuracy)
                    out2[o] = g * xtbuf[o];      // b_t
                }
            }
        }
    }
}

// s_t = a_t * s_{t-1} + b_t per (n,c) channel; writes bf16 (consumed as WMMA A).
__global__ __launch_bounds__(256) void scan_kernel(
    const float* __restrict__ a, const float* __restrict__ b,
    unsigned short* __restrict__ sbf) {
    int ch = blockIdx.x * 256 + threadIdx.x;   // 0 .. N*C-1
    int n  = ch / Cc;
    int c  = ch - n * Cc;
    size_t base = (size_t)n * Tt * Cc + c;
    float st = 0.0f;
#pragma unroll 4
    for (int t = 0; t < Tt; ++t) {
        size_t o = base + (size_t)t * Cc;
        st = fmaf(a[o], st, b[o]);
        sbf[o] = f2bf(st);
    }
}

// bulk f32 -> bf16 conversion (4 elems/thread, vectorized)
__global__ __launch_bounds__(256) void cvt_kernel(
    const float* __restrict__ in, unsigned short* __restrict__ out, int n) {
    int i = (blockIdx.x * 256 + threadIdx.x) * 4;
    if (i >= n) return;
    float4 f = *reinterpret_cast<const float4*>(in + i);
    uint2 pk;
    pk.x = (unsigned)f2bf(f.x) | ((unsigned)f2bf(f.y) << 16);
    pk.y = (unsigned)f2bf(f.z) | ((unsigned)f2bf(f.w) << 16);
    *reinterpret_cast<uint2*>(out + i) = pk;
}

extern "C" void kernel_launch(void* const* d_in, const int* in_sizes, int n_in,
                              void* d_out, int out_size, void* d_ws, size_t ws_size,
                              hipStream_t stream) {
    const float* x_seq   = (const float*)d_in[0];
    const float* sup_seq = (const float*)d_in[1];
    const float* W_in    = (const float*)d_in[2];
    const float* W_out   = (const float*)d_in[3];
    const float* W_bg    = (const float*)d_in[4];
    const float* b_bg    = (const float*)d_in[5];
    const float* W_sg    = (const float*)d_in[6];
    const float* b_sg    = (const float*)d_in[7];
    const float* W_f1    = (const float*)d_in[8];
    const float* b_f1    = (const float*)d_in[9];
    const float* W_f2    = (const float*)d_in[10];
    const float* b_f2    = (const float*)d_in[11];
    float* out = (float*)d_out;

    // ---- workspace layout (byte offsets; aggressive buffer reuse) ----
    const size_t MiB = 1u << 20;
    unsigned char* ws = (unsigned char*)d_ws;
    unsigned short* wbf_in  = (unsigned short*)(ws);                 // bf16 weights
    unsigned short* wbf_bg  = wbf_in + (size_t)Cc * Cc;
    unsigned short* wbf_out = wbf_in + 2 * (size_t)Cc * Cc;
    unsigned short* wbf_f1  = wbf_in + 3 * (size_t)Cc * Cc;
    unsigned short* wbf_f2  = wbf_f1 + (size_t)Cc * Hh;              // ends < 16 MiB
    unsigned short* xbf  = (unsigned short*)(ws + 16 * MiB);   // 24 MiB, dead after G1
    unsigned short* Xbf  = (unsigned short*)(ws + 40 * MiB);   // 24 MiB, dead after G2
    float*          Xf   = (float*)         (ws + 64 * MiB);   // 48 MiB, dead after G3
    unsigned short* Hbf  = (unsigned short*)(ws + 112 * MiB);  // 48 MiB, dead after G3
    float*          XTf  = (float*)         (ws + 160 * MiB);  // 48 MiB, dead after G4
    unsigned short* XTbf = (unsigned short*)(ws + 208 * MiB);  // 24 MiB, dead after G4
    float*          Af   = (float*)         (ws + 16 * MiB);   // reuse xbf+Xbf region
    float*          Bf   = (float*)         (ws + 64 * MiB);   // reuse Xf
    unsigned short* Sbf  = (unsigned short*)(ws + 112 * MiB);  // reuse Hbf

    dim3 blk(256);
    dim3 gC(Mm / 128, Cc / 256);   // 128 x 3
    dim3 gH(Mm / 128, Hh / 256);   // 128 x 6

    // ---- one-shot f32 -> bf16 conversions (weights + input) ----
    cvt_kernel<<<(Cc * Cc) / 1024, blk, 0, stream>>>(W_in,  wbf_in,  Cc * Cc);
    cvt_kernel<<<(Cc * Cc) / 1024, blk, 0, stream>>>(W_bg,  wbf_bg,  Cc * Cc);
    cvt_kernel<<<(Cc * Cc) / 1024, blk, 0, stream>>>(W_out, wbf_out, Cc * Cc);
    cvt_kernel<<<(Cc * Hh) / 1024, blk, 0, stream>>>(W_f1,  wbf_f1,  Cc * Hh);
    cvt_kernel<<<(Cc * Hh) / 1024, blk, 0, stream>>>(W_f2,  wbf_f2,  Cc * Hh);
    cvt_kernel<<<(Mm * Cc) / 1024, blk, 0, stream>>>(x_seq, xbf,     Mm * Cc);

    // 1) X = x @ W_in^T            (f32 + bf16 copies)
    gemm_bf16_wmma<EPI_XDUAL><<<gC, blk, 0, stream>>>(
        xbf, wbf_in, Xf, Xbf, nullptr, nullptr, nullptr, nullptr,
        nullptr, nullptr, nullptr, Cc, Cc);
    // 2) H = gelu(X @ W_f1^T + b)  (bf16)
    gemm_bf16_wmma<EPI_GELU><<<gH, blk, 0, stream>>>(
        Xbf, wbf_f1, nullptr, Hbf, nullptr, b_f1, nullptr, nullptr,
        nullptr, nullptr, nullptr, Cc, Hh);
    // 3) XT = X + H @ W_f2^T + b   (f32 + bf16)
    gemm_bf16_wmma<EPI_RESID_DUAL><<<gC, blk, 0, stream>>>(
        Hbf, wbf_f2, XTf, XTbf, nullptr, b_f2, Xf, nullptr,
        nullptr, nullptr, nullptr, Hh, Cc);
    // 4) gate: a = 1-g, b = g*XT
    gemm_bf16_wmma<EPI_GATE><<<gC, blk, 0, stream>>>(
        XTbf, wbf_bg, Af, nullptr, Bf, b_bg, nullptr, sup_seq,
        W_sg, b_sg, XTf, Cc, Cc);
    // 5) linear recurrence -> S (bf16)
    scan_kernel<<<(Nb * Cc) / 256, blk, 0, stream>>>(Af, Bf, Sbf);
    // 6) out = S @ W_out^T (f32)
    gemm_bf16_wmma<EPI_OUT><<<gC, blk, 0, stream>>>(
        Sbf, wbf_out, out, nullptr, nullptr, nullptr, nullptr, nullptr,
        nullptr, nullptr, nullptr, Cc, Cc);
    (void)in_sizes; (void)n_in; (void)out_size; (void)ws_size;
}
